// ExplicitIntegralGNN_88510686036722
// MI455X (gfx1250) — compile-verified, matrix operand order checked by reading, not run on hardware
//
#include <hip/hip_runtime.h>
#include <math.h>

// ---------------------------------------------------------------------------
// CDNA5 / gfx1250 GAT pipeline. wave32 everywhere. Dense projections use
// V_WMMA_F32_16X16X32_BF16 (compute-bound: ~84 GFLOP of 1024x1024 projections
// over 20000 nodes; bf16 WMMA is the high-throughput CDNA5 path, f32 acc).
// Edge aggregation is bandwidth/atomic bound (~1.5 GB/layer at 23.3 TB/s).
// Workspace layout (~370 MB):
//   nb0 (agg), nb1, nb2 : N*1024 f32 node buffers (ping-pong + accumulator)
//   proj                : N*1024 f32 projected features
//   normbf              : N*1024 bf16 LayerNorm output (WMMA A operand)
//   wbf                 : 1024*1024 bf16 weight cast
//   als/ald/emax/denom  : N*8 f32 attention scalars
//   exv                 : (E+N)*8 f32 per-edge exp scores
// ---------------------------------------------------------------------------

typedef __attribute__((ext_vector_type(16))) __bf16 v16bf;
typedef __attribute__((ext_vector_type(8)))  float  v8f;

__device__ __forceinline__ float wave_sum(float v) {
  #pragma unroll
  for (int m = 16; m > 0; m >>= 1) v += __shfl_xor(v, m, 32);
  return v;
}

__device__ __forceinline__ void atomic_max_f32(float* addr, float val) {
  if (val >= 0.0f) atomicMax((int*)addr, __float_as_int(val));
  else             atomicMin((unsigned int*)addr, __float_as_uint(val));
}

__device__ __forceinline__ void edge_nodes(const int* __restrict__ ei, int e,
                                           int E, int& s, int& d) {
  if (e < E) { s = ei[e]; d = ei[E + e]; }
  else       { s = e - E; d = e - E; }        // self loops appended
}

// --- LayerNorm (biased var, eps 1e-6) -> bf16 ------------------------------
__global__ void k_layernorm(const float* __restrict__ X,
                            const float* __restrict__ g,
                            const float* __restrict__ b,
                            __bf16* __restrict__ out, int Nrows, int C) {
  int wid  = (blockIdx.x * blockDim.x + threadIdx.x) >> 5;
  int lane = threadIdx.x & 31;
  if (wid >= Nrows) return;
  const float* row = X + (size_t)wid * C;
  float s = 0.f, s2 = 0.f;
  for (int c = lane; c < C; c += 32) { float v = row[c]; s += v; s2 += v * v; }
  s = wave_sum(s); s2 = wave_sum(s2);
  float mu  = s / (float)C;
  float var = s2 / (float)C - mu * mu;
  float inv = rsqrtf(var + 1e-6f);
  __bf16* orow = out + (size_t)wid * C;
  for (int c = lane; c < C; c += 32) {
    float v = (row[c] - mu) * inv * g[c] + b[c];
    orow[c] = (__bf16)v;
  }
}

// --- f32 -> bf16 cast ------------------------------------------------------
__global__ void k_cast_bf16(const float* __restrict__ in,
                            __bf16* __restrict__ out, int n) {
  int i = blockIdx.x * blockDim.x + threadIdx.x;
  if (i < n) out[i] = (__bf16)in[i];
}

// --- WMMA GEMM: C[N,O] = A[N,K](bf16) * W[O,K](bf16)^T, f32 accumulate -----
// wave tile 32x64 (2x4 wmma), block = 4 waves -> 32x256 block tile.
// K % 32 == 0, O % 256 == 0 (layers 0-2: O = 1024).
// A rows are CLAMPED (not branched) so the K-loop is a pure load+WMMA stream;
// out-of-range rows compute garbage but stores are row-guarded.
__global__ __launch_bounds__(128)
void k_gemm_wmma_bf16(const __bf16* __restrict__ A, const __bf16* __restrict__ W,
                      float* __restrict__ C, int Nrows, int K, int O) {
  const int lane = threadIdx.x & 31;
  const int wave = threadIdx.x >> 5;
  const int m0   = blockIdx.x * 32;
  const int n0   = blockIdx.y * 256 + wave * 64;
  const int half = lane >> 4;       // 0: K 0..7 & 16..23 ; 1: K 8..15 & 24..31
  const int l    = lane & 15;
  const int kb   = half * 8;

  // Hoisted, branchless base pointers.
  const __bf16* ap[2];
  #pragma unroll
  for (int mt = 0; mt < 2; ++mt) {
    int row = m0 + mt * 16 + l;
    row = row < Nrows ? row : Nrows - 1;           // clamp, no EXEC branching
    ap[mt] = A + (size_t)row * K + kb;
  }
  const __bf16* bp[4];
  #pragma unroll
  for (int nt = 0; nt < 4; ++nt)
    bp[nt] = W + (size_t)(n0 + nt * 16 + l) * K + kb;

  v8f acc[2][4];
  #pragma unroll
  for (int mt = 0; mt < 2; ++mt)
    #pragma unroll
    for (int nt = 0; nt < 4; ++nt) acc[mt][nt] = (v8f)0.0f;

  for (int k0 = 0; k0 < K; k0 += 32) {
    v16bf a[2];
    #pragma unroll
    for (int mt = 0; mt < 2; ++mt) {
      const __bf16* p = ap[mt] + k0;
      __builtin_prefetch(p + 32, 0, 1);            // global_prefetch_b8
      #pragma unroll
      for (int j = 0; j < 8; ++j) { a[mt][j] = p[j]; a[mt][8 + j] = p[16 + j]; }
    }
    v16bf bfr[4];
    #pragma unroll
    for (int nt = 0; nt < 4; ++nt) {
      const __bf16* p = bp[nt] + k0;
      #pragma unroll
      for (int j = 0; j < 8; ++j) { bfr[nt][j] = p[j]; bfr[nt][8 + j] = p[16 + j]; }
    }
    #pragma unroll
    for (int mt = 0; mt < 2; ++mt)
      #pragma unroll
      for (int nt = 0; nt < 4; ++nt)
        acc[mt][nt] = __builtin_amdgcn_wmma_f32_16x16x32_bf16(
            false, a[mt], false, bfr[nt], (short)0, acc[mt][nt], false, false);
  }
  // C/D layout: lanes 0-15 hold M = vgpr (0..7), lanes 16-31 hold M = vgpr+8
  #pragma unroll
  for (int mt = 0; mt < 2; ++mt)
    #pragma unroll
    for (int nt = 0; nt < 4; ++nt) {
      int col = n0 + nt * 16 + l;
      #pragma unroll
      for (int j = 0; j < 8; ++j) {
        int row = m0 + mt * 16 + j + half * 8;
        if (row < Nrows) C[(size_t)row * O + col] = acc[mt][nt][j];
      }
    }
}

// --- tiny GEMM for layer 3 (O = 3): wave per node --------------------------
__global__ void k_gemm_small(const __bf16* __restrict__ A,
                             const float* __restrict__ W,
                             float* __restrict__ C, int Nrows, int K, int O) {
  int wid  = (blockIdx.x * blockDim.x + threadIdx.x) >> 5;
  int lane = threadIdx.x & 31;
  if (wid >= Nrows) return;
  const __bf16* arow = A + (size_t)wid * K;
  float acc[4] = {0.f, 0.f, 0.f, 0.f};
  for (int k = lane; k < K; k += 32) {
    float av = (float)arow[k];
    for (int o = 0; o < O; ++o) acc[o] += av * W[(size_t)o * K + k];
  }
  for (int o = 0; o < O; ++o) {
    float r = wave_sum(acc[o]);
    if (lane == 0) C[(size_t)wid * O + o] = r;
  }
}

// --- per-layer init: agg = 0, emax = -inf, denom = 0 -----------------------
__global__ void k_init(float* __restrict__ agg, float* __restrict__ emax,
                       float* __restrict__ denom, int aggn, int nh) {
  int i = blockIdx.x * blockDim.x + threadIdx.x;
  if (i < aggn) agg[i] = 0.0f;
  if (i < nh) { emax[i] = -INFINITY; denom[i] = 0.0f; }
}

// --- attention coefficients: al[n,h] = <proj[n,h,:], a[h,:]> ---------------
__global__ void k_attn(const float* __restrict__ proj,
                       const float* __restrict__ asrc,
                       const float* __restrict__ adst,
                       float* __restrict__ als, float* __restrict__ ald,
                       int Nrows, int H, int C) {
  int wid  = (blockIdx.x * blockDim.x + threadIdx.x) >> 5;
  int lane = threadIdx.x & 31;
  if (wid >= Nrows * H) return;
  int n = wid / H, h = wid % H;
  const float* hp = proj + (size_t)n * (H * C) + h * C;
  float ss = 0.f, sd = 0.f;
  for (int c = lane; c < C; c += 32) {
    float v = hp[c];
    ss += v * asrc[h * C + c];
    sd += v * adst[h * C + c];
  }
  ss = wave_sum(ss); sd = wave_sum(sd);
  if (lane == 0) { als[wid] = ss; ald[wid] = sd; }
}

// --- segment softmax pass 1: atomic max over dst ---------------------------
__global__ void k_edge_max(const int* __restrict__ ei, int E, int ET, int H,
                           const float* __restrict__ als,
                           const float* __restrict__ ald,
                           float* __restrict__ emax) {
  int idx = blockIdx.x * blockDim.x + threadIdx.x;
  if (idx >= ET * H) return;
  int e = idx / H, h = idx % H;
  int s, d; edge_nodes(ei, e, E, s, d);
  float sc = als[s * H + h] + ald[d * H + h];
  sc = sc > 0.0f ? sc : 0.2f * sc;                 // leaky_relu(0.2)
  atomic_max_f32(&emax[d * H + h], sc);
}

// --- pass 2: ex = exp(score - max), denom += ex ----------------------------
__global__ void k_edge_exp(const int* __restrict__ ei, int E, int ET, int H,
                           const float* __restrict__ als,
                           const float* __restrict__ ald,
                           const float* __restrict__ emax,
                           float* __restrict__ denom,
                           float* __restrict__ exv) {
  int idx = blockIdx.x * blockDim.x + threadIdx.x;
  if (idx >= ET * H) return;
  int e = idx / H, h = idx % H;
  int s, d; edge_nodes(ei, e, E, s, d);
  float sc = als[s * H + h] + ald[d * H + h];
  sc = sc > 0.0f ? sc : 0.2f * sc;
  float ex = __expf(sc - emax[d * H + h]);
  exv[idx] = ex;
  atomicAdd(&denom[d * H + h], ex);
}

// --- pass 3: agg[dst,h,:] += alpha * proj[src,h,:] (wave per edge-head) ----
__global__ void k_aggregate(const int* __restrict__ ei, int E, int ET, int H,
                            int C, const float* __restrict__ exv,
                            const float* __restrict__ denom,
                            const float* __restrict__ proj,
                            float* __restrict__ agg) {
  int wid  = (blockIdx.x * blockDim.x + threadIdx.x) >> 5;
  int lane = threadIdx.x & 31;
  if (wid >= ET * H) return;
  int e = wid / H, h = wid % H;
  int s, d; edge_nodes(ei, e, E, s, d);
  float alpha = exv[wid] / (denom[d * H + h] + 1e-16f);
  const float* hp = proj + (size_t)s * (H * C) + h * C;
  float*       op = agg  + (size_t)d * (H * C) + h * C;
  for (int c = lane; c < C; c += 32)
    atomicAdd(&op[c], alpha * hp[c]);
}

// --- epilogue layers 0-2: gelu(agg + bias) (+ x_res) -----------------------
__global__ void k_epilogue(const float* __restrict__ agg,
                           const float* __restrict__ bias,
                           const float* __restrict__ xres, int add_res,
                           float* __restrict__ out, int Nrows, int HC) {
  int idx = blockIdx.x * blockDim.x + threadIdx.x;
  if (idx >= Nrows * HC) return;
  float v = agg[idx] + bias[idx % HC];
  v = 0.5f * v * (1.0f + erff(v * 0.70710678118654752f));   // exact-erf gelu
  if (add_res) v += xres[idx];
  out[idx] = v;
}

// --- final: out = agg3 + bias3 + x @ res_w^T + res_b (wave per node) -------
__global__ void k_final(const float* __restrict__ agg3,
                        const float* __restrict__ bias3,
                        const float* __restrict__ x,
                        const float* __restrict__ res_w,
                        const float* __restrict__ res_b,
                        float* __restrict__ out, int Nrows) {
  int wid  = (blockIdx.x * blockDim.x + threadIdx.x) >> 5;
  int lane = threadIdx.x & 31;
  if (wid >= Nrows) return;
  float x0 = x[(size_t)wid * 64 + lane];
  float x1 = x[(size_t)wid * 64 + 32 + lane];
  for (int c = 0; c < 3; ++c) {
    float dp = x0 * res_w[c * 64 + lane] + x1 * res_w[c * 64 + 32 + lane];
    dp = wave_sum(dp);
    if (lane == 0)
      out[(size_t)wid * 3 + c] = agg3[(size_t)wid * 3 + c] + bias3[c] + dp + res_b[c];
  }
}

// ---------------------------------------------------------------------------
extern "C" void kernel_launch(void* const* d_in, const int* in_sizes, int n_in,
                              void* d_out, int out_size, void* d_ws, size_t ws_size,
                              hipStream_t stream) {
  const float* x  = (const float*)d_in[0];
  const int*   ei = (const int*)d_in[1];
  const int N  = in_sizes[0] / 64;
  const int E  = in_sizes[1] / 2;
  const int ET = E + N;

  const float *ln_g[4], *ln_b[4], *Wp[4], *asrc[4], *adst[4], *bias[4];
  for (int i = 0; i < 4; ++i) {
    ln_g[i] = (const float*)d_in[2 + 6 * i + 0];
    ln_b[i] = (const float*)d_in[2 + 6 * i + 1];
    Wp[i]   = (const float*)d_in[2 + 6 * i + 2];
    asrc[i] = (const float*)d_in[2 + 6 * i + 3];
    adst[i] = (const float*)d_in[2 + 6 * i + 4];
    bias[i] = (const float*)d_in[2 + 6 * i + 5];
  }
  const float* res_w = (const float*)d_in[26];
  const float* res_b = (const float*)d_in[27];

  char* wsb = (char*)d_ws;
  size_t off = 0;
  auto alloc = [&](size_t bytes) -> void* {
    void* p = wsb + off;
    off = (off + bytes + 255) & ~(size_t)255;
    return p;
  };
  const size_t NF = (size_t)N * 1024;
  float*  nb0    = (float*)alloc(NF * 4);          // agg accumulator
  float*  nb1    = (float*)alloc(NF * 4);
  float*  nb2    = (float*)alloc(NF * 4);
  float*  proj   = (float*)alloc(NF * 4);
  __bf16* normbf = (__bf16*)alloc(NF * 2);
  __bf16* wbf    = (__bf16*)alloc((size_t)1024 * 1024 * 2);
  float*  als    = (float*)alloc((size_t)N * 8 * 4);
  float*  ald    = (float*)alloc((size_t)N * 8 * 4);
  float*  emax   = (float*)alloc((size_t)N * 8 * 4);
  float*  denom  = (float*)alloc((size_t)N * 8 * 4);
  float*  exv    = (float*)alloc((size_t)ET * 8 * 4);
  (void)ws_size; (void)n_in; (void)out_size;

  const int cin[4]  = {64, 1024, 1024, 1024};
  const int cout[4] = {128, 128, 128, 3};
  const int Hh[4]   = {8, 8, 8, 1};
  const int HC[4]   = {1024, 1024, 1024, 3};
  const float* inbuf[4] = {x, nb1, nb2, nb1};
  float* nextbuf[3]     = {nb1, nb2, nb1};
  const int add_res[3]  = {0, 1, 1};

  for (int i = 0; i < 4; ++i) {
    // LayerNorm -> bf16 (wave per row)
    k_layernorm<<<(N + 7) / 8, 256, 0, stream>>>(inbuf[i], ln_g[i], ln_b[i],
                                                 normbf, N, cin[i]);
    // Projection
    if (i < 3) {
      int wn = HC[i] * cin[i];
      k_cast_bf16<<<(wn + 255) / 256, 256, 0, stream>>>(Wp[i], wbf, wn);
      dim3 grid((N + 31) / 32, HC[i] / 256);
      k_gemm_wmma_bf16<<<grid, 128, 0, stream>>>(normbf, wbf, proj, N, cin[i], HC[i]);
    } else {
      k_gemm_small<<<(N + 7) / 8, 256, 0, stream>>>(normbf, Wp[i], proj, N,
                                                    cin[i], HC[i]);
    }
    // Init accumulators
    int aggn = N * HC[i], nh = N * Hh[i];
    k_init<<<(aggn + 255) / 256, 256, 0, stream>>>(nb0, emax, denom, aggn, nh);
    // Attention coefficients (wave per node-head)
    k_attn<<<(N * Hh[i] + 7) / 8, 256, 0, stream>>>(proj, asrc[i], adst[i],
                                                    als, ald, N, Hh[i], cout[i]);
    // Segment softmax + aggregation
    int eh = ET * Hh[i];
    k_edge_max<<<(eh + 255) / 256, 256, 0, stream>>>(ei, E, ET, Hh[i], als, ald, emax);
    k_edge_exp<<<(eh + 255) / 256, 256, 0, stream>>>(ei, E, ET, Hh[i], als, ald,
                                                     emax, denom, exv);
    k_aggregate<<<(eh + 7) / 8, 256, 0, stream>>>(ei, E, ET, Hh[i], cout[i],
                                                  exv, denom, proj, nb0);
    // Epilogue
    if (i < 3) {
      k_epilogue<<<(N * HC[i] + 255) / 256, 256, 0, stream>>>(
          nb0, bias[i], inbuf[i], add_res[i], nextbuf[i], N, HC[i]);
    }
  }
  k_final<<<(N + 7) / 8, 256, 0, stream>>>(nb0, bias[3], x, res_w, res_b,
                                           (float*)d_out, N);
}
